// MultiScaleFeatureFusion_85633057947801
// MI455X (gfx1250) — matrix operand order repaired and checked by reading, not compiled, not run on hardware
//
#include <hip/hip_runtime.h>

typedef __attribute__((ext_vector_type(16))) _Float16 v16h;
typedef __attribute__((ext_vector_type(8)))  float    v8f;

static constexpr int N = 4096;   // 64*64 spatial positions
static constexpr int NB = 4;     // batch
#define LOG2E 1.44269504088896340736f

// ---------------------------------------------------------------------------
// Bilinear upsample (align_corners=True) to 64x64.  out[b][c][p], p = y*64+x
// ---------------------------------------------------------------------------
__global__ __launch_bounds__(256) void upsample_kernel(
    const float* __restrict__ in, float* __restrict__ out, int C, int IH, int IW)
{
    int b   = blockIdx.y;
    int idx = blockIdx.x * 256 + threadIdx.x;      // over C*N
    int c   = idx >> 12;
    int p   = idx & (N - 1);
    int y   = p >> 6, x = p & 63;
    float sy = (IH - 1) * (1.0f / 63.0f);
    float sx = (IW - 1) * (1.0f / 63.0f);
    float cy = y * sy, cx = x * sx;
    int iy0 = min((int)cy, IH - 2);
    int ix0 = min((int)cx, IW - 2);
    float wy = cy - (float)iy0, wx = cx - (float)ix0;
    const float* bp = in + ((size_t)b * C + c) * (size_t)(IH * IW);
    float v00 = bp[iy0 * IW + ix0];
    float v01 = bp[iy0 * IW + ix0 + 1];
    float v10 = bp[(iy0 + 1) * IW + ix0];
    float v11 = bp[(iy0 + 1) * IW + ix0 + 1];
    float r0 = v00 * (1.f - wy) + v10 * wy;
    float r1 = v01 * (1.f - wy) + v11 * wy;
    out[((size_t)b * C + c) * N + p] = r0 * (1.f - wx) + r1 * wx;
}

// ---------------------------------------------------------------------------
// Fold the value path: wvav[j] = sum_c wa[c]*wv[c][j],  bva = wa . bv
// ---------------------------------------------------------------------------
__global__ void prep_wvav(const float* __restrict__ wa, const float* __restrict__ wv,
                          const float* __restrict__ bv, float* __restrict__ wvav,
                          float* __restrict__ bva, int C)
{
    int j = blockIdx.x * 128 + threadIdx.x;
    if (j < C) {
        float s = 0.f;
        for (int c = 0; c < C; ++c) s += wa[c] * wv[(size_t)c * C + j];
        wvav[j] = s;
    }
    if (blockIdx.x == 0 && threadIdx.x == 0) {
        float s = 0.f;
        for (int c = 0; c < C; ++c) s += wa[c] * bv[c];
        *bva = s;
    }
}

__global__ void transpose_w1(const float* __restrict__ w1, float* __restrict__ w1t)
{
    int idx = blockIdx.x * 256 + threadIdx.x;      // over 32*896
    if (idx < 32 * 896) {
        int o = idx / 896, c = idx - o * 896;
        w1t[c * 32 + o] = w1[idx];
    }
}

// ---------------------------------------------------------------------------
// WMMA q/k projection.
//   qT[b][n][t] = log2(e) * ( sum_c wq[t][c]*x[b][c][n] + bq[t] )   (f16)
//   kT[b][n][t] =            sum_c wk[t][c]*x[b][c][n] + bk[t]      (f16)
// One block = one 16-position tile.  xT tile (16n x 32c, f16) staged in LDS
// per K-step; 2*(D/16) waves each own one 16x16 output tile (q tiles then k
// tiles); K accumulated with v_wmma_f32_16x16x32_f16 (A = xT from LDS,
// B = w f32->f16 in-register).  D < DPAD pad region (scale 2) zero-filled.
// ---------------------------------------------------------------------------
template<int C, int D, int DPAD>
__global__ __launch_bounds__(4 * D) void proj_qk_wmma(
    const float* __restrict__ x,
    const float* __restrict__ wq, const float* __restrict__ bq,
    const float* __restrict__ wk, const float* __restrict__ bk,
    _Float16* __restrict__ qT, _Float16* __restrict__ kT)
{
    constexpr int TW = D / 16;          // t-tiles per matrix
    constexpr int NW = 2 * TW;          // waves per block
    __shared__ _Float16 xs[16][34];     // padded: (n*17 + c/2) % 64 distinct

    int b     = blockIdx.y;
    int nbase = blockIdx.x * 16;
    int wave  = threadIdx.x >> 5;
    int lane  = threadIdx.x & 31;
    int which = wave >= TW;             // 0 = q, 1 = k
    int tbase = (which ? wave - TW : wave) * 16;
    const float* w    = which ? wk : wq;
    const float* bias = which ? bk : bq;
    const float* xb   = x + (size_t)b * C * N;

    int hi    = lane >> 4;
    int row   = lane & 15;              // A row (position) / D column index
    int col   = row;                    // B column (t) for this lane
    int koffA = hi * 8;
    int koffB = hi * 16;

    v8f acc = {};
    for (int ck = 0; ck < C; ck += 32) {
        __syncthreads();
        for (int i = threadIdx.x; i < 32 * 16; i += NW * 32) {
            int nn = i & 15;
            int cc = i >> 4;
            xs[nn][cc] = (_Float16)xb[(size_t)(ck + cc) * N + nbase + nn];
        }
        __syncthreads();

        v16h a;
        #pragma unroll
        for (int j = 0; j < 8; ++j) {
            a[j]     = xs[row][koffA + j];
            a[8 + j] = xs[row][16 + koffA + j];
        }
        const float* wr = w + (size_t)(tbase + col) * C + ck;
        v16h bf;
        #pragma unroll
        for (int j = 0; j < 16; ++j) bf[j] = (_Float16)wr[(j < 8 ? 0 : 8) + koffB / 2 + j];

        acc = __builtin_amdgcn_wmma_f32_16x16x32_f16(
                false, a, false, bf, (short)0, acc, false, false);
    }

    float bs    = bias[tbase + col];
    float scale = which ? 1.0f : LOG2E;          // fold log2(e) into q
    _Float16* out = (which ? kT : qT) + (size_t)b * N * DPAD;
    #pragma unroll
    for (int r = 0; r < 8; ++r) {
        int n = nbase + hi * 8 + r;
        out[(size_t)n * DPAD + tbase + col] = (_Float16)((acc[r] + bs) * scale);
        if (D < DPAD)                            // zero pad dims [D, DPAD)
            out[(size_t)n * DPAD + D + tbase + col] = (_Float16)0.f;
    }
}

// u[b][n] = wvav . x[b][:,n] + bva
template<int C>
__global__ __launch_bounds__(256) void proj_u_kernel(
    const float* __restrict__ x, const float* __restrict__ wvav,
    const float* __restrict__ bva, float* __restrict__ uo)
{
    int b = blockIdx.y;
    int n = blockIdx.x * 256 + threadIdx.x;
    const float* xb = x + (size_t)b * C * N + n;
    float s = *bva;
    #pragma unroll 4
    for (int c = 0; c < C; ++c) s += wvav[c] * xb[(size_t)c * N];
    uo[b * N + n] = s;
}

// ---------------------------------------------------------------------------
// Flash-style attention with SCALAR values (q pre-scaled by log2e -> exp2):
//   a[n] = sum_m softmax_m(q_n . k_m) * u[m] + ba
// One wave owns a 16-query tile; loops 256 key tiles; logits via
// v_wmma_f32_16x16x32_f16 (DPAD/32 fragments); online softmax per row held
// across 16 lanes, combined with width-16 xor shuffles at the end.
// ---------------------------------------------------------------------------
template<int DPAD>
__global__ __launch_bounds__(256) void attn_kernel(
    const _Float16* __restrict__ qT, const _Float16* __restrict__ kT,
    const float* __restrict__ u, const float* __restrict__ ba,
    float* __restrict__ a_out)
{
    constexpr int NFRAG = DPAD / 32;
    int b     = blockIdx.y;
    int wave  = threadIdx.x >> 5;
    int lane  = threadIdx.x & 31;
    int qtile = blockIdx.x * 8 + wave;
    int qbase = qtile * 16;
    const _Float16* qb = qT + (size_t)b * N * DPAD;
    const _Float16* kb = kT + (size_t)b * N * DPAD;
    const float*    ub = u + b * N;

    int half16 = lane >> 4;
    int row    = lane & 15;
    int koffA  = half16 * 8;

    v16h afrag[NFRAG];
    const _Float16* qrow = qb + (size_t)(qbase + row) * DPAD;
    #pragma unroll
    for (int f = 0; f < NFRAG; ++f) {
        #pragma unroll
        for (int j = 0; j < 8; ++j) {
            afrag[f][j]     = qrow[f * 32 + koffA + j];
            afrag[f][8 + j] = qrow[f * 32 + 16 + koffA + j];
        }
    }

    float mx[8], lsum[8], acc[8];
    #pragma unroll
    for (int r = 0; r < 8; ++r) { mx[r] = -1e30f; lsum[r] = 0.f; acc[r] = 0.f; }

    int col   = lane & 15;
    int koffB = half16 * 16;
    for (int kt = 0; kt < N / 16; ++kt) {
        int kbase = kt * 16;
        const _Float16* krow = kb + (size_t)(kbase + col) * DPAD;
        if (kt + 1 < N / 16)
            __builtin_prefetch(kb + (size_t)(kbase + 16 + col) * DPAD, 0, 3);

        v8f c8 = {};
        #pragma unroll
        for (int f = 0; f < NFRAG; ++f) {
            v16h bfrag;
            #pragma unroll
            for (int j = 0; j < 16; ++j) bfrag[j] = krow[f * 32 + (j < 8 ? 0 : 8) + koffB / 2 + j];
            c8 = __builtin_amdgcn_wmma_f32_16x16x32_f16(
                    false, afrag[f], false, bfrag, (short)0, c8, false, false);
        }
        float uval = ub[kbase + col];
        #pragma unroll
        for (int r = 0; r < 8; ++r) {
            float s    = c8[r];                  // already in log2 domain
            float mnew = fmaxf(mx[r], s);
            float corr = exp2f(mx[r] - mnew);
            float e    = exp2f(s - mnew);
            lsum[r] = lsum[r] * corr + e;
            acc[r]  = acc[r]  * corr + e * uval;
            mx[r]   = mnew;
        }
    }

    #pragma unroll
    for (int off = 1; off < 16; off <<= 1) {
        #pragma unroll
        for (int r = 0; r < 8; ++r) {
            float m2   = __shfl_xor(mx[r],   off, 16);
            float l2   = __shfl_xor(lsum[r], off, 16);
            float a2   = __shfl_xor(acc[r],  off, 16);
            float mnew = fmaxf(mx[r], m2);
            float e1   = exp2f(mx[r] - mnew);
            float e2   = exp2f(m2    - mnew);
            lsum[r] = lsum[r] * e1 + l2 * e2;
            acc[r]  = acc[r]  * e1 + a2 * e2;
            mx[r]   = mnew;
        }
    }
    float bav = *ba;
    if (col == 0) {
        int rbase = qbase + half16 * 8;
        #pragma unroll
        for (int r = 0; r < 8; ++r)
            a_out[b * N + rbase + r] = acc[r] / lsum[r] + bav;
    }
}

// ---------------------------------------------------------------------------
// Fused head: prod = a0*a1*a2; t[o] = w1[o,:].xcat[:,n]; relu; w2 reduce.
// ---------------------------------------------------------------------------
__global__ __launch_bounds__(256) void final_kernel(
    const float* __restrict__ x3, const float* __restrict__ x2u,
    const float* __restrict__ x1u,
    const float* __restrict__ a0, const float* __restrict__ a1,
    const float* __restrict__ a2,
    const float* __restrict__ w1t, const float* __restrict__ b1,
    const float* __restrict__ w2,  const float* __restrict__ b2,
    float* __restrict__ out)
{
    __shared__ float xs[8][896];
    int b     = blockIdx.y;
    int nbase = blockIdx.x * 8;
    for (int i = threadIdx.x; i < 8 * 896; i += 256) {
        int p = i & 7;
        int c = i >> 3;
        int n = nbase + p;
        float v;
        if (c < 128)      v = x3 [((size_t)b * 128 +  c       ) * N + n];
        else if (c < 384) v = x2u[((size_t)b * 256 + (c - 128)) * N + n];
        else              v = x1u[((size_t)b * 512 + (c - 384)) * N + n];
        xs[p][c] = v;
    }
    __syncthreads();

    int o = threadIdx.x & 31;
    int p = threadIdx.x >> 5;
    int n = nbase + p;
    float t = 0.f;
    #pragma unroll 4
    for (int c = 0; c < 896; ++c) t += w1t[c * 32 + o] * xs[p][c];
    float prod = a0[b * N + n] * a1[b * N + n] * a2[b * N + n];
    float h = fmaxf(prod * t + b1[o], 0.f);
    float contrib = w2[o] * h;
    #pragma unroll
    for (int off = 16; off > 0; off >>= 1)
        contrib += __shfl_xor(contrib, off, 32);
    if (o == 0) out[b * N + n] = contrib + *b2;
}

// ---------------------------------------------------------------------------
extern "C" void kernel_launch(void* const* d_in, const int* in_sizes, int n_in,
                              void* d_out, int out_size, void* d_ws, size_t ws_size,
                              hipStream_t stream)
{
    (void)in_sizes; (void)n_in; (void)out_size; (void)ws_size;
    const float* x3  = (const float*)d_in[0];
    const float* x2  = (const float*)d_in[1];
    const float* x1  = (const float*)d_in[2];
    const float* wq0 = (const float*)d_in[3];  const float* bq0 = (const float*)d_in[4];
    const float* wk0 = (const float*)d_in[5];  const float* bk0 = (const float*)d_in[6];
    const float* wv0 = (const float*)d_in[7];  const float* bv0 = (const float*)d_in[8];
    const float* wa0 = (const float*)d_in[9];  const float* ba0 = (const float*)d_in[10];
    const float* wq1 = (const float*)d_in[11]; const float* bq1 = (const float*)d_in[12];
    const float* wk1 = (const float*)d_in[13]; const float* bk1 = (const float*)d_in[14];
    const float* wv1 = (const float*)d_in[15]; const float* bv1 = (const float*)d_in[16];
    const float* wa1 = (const float*)d_in[17]; const float* ba1 = (const float*)d_in[18];
    const float* wq2 = (const float*)d_in[19]; const float* bq2 = (const float*)d_in[20];
    const float* wk2 = (const float*)d_in[21]; const float* bk2 = (const float*)d_in[22];
    const float* wv2 = (const float*)d_in[23]; const float* bv2 = (const float*)d_in[24];
    const float* wa2 = (const float*)d_in[25]; const float* ba2 = (const float*)d_in[26];
    const float* w1  = (const float*)d_in[27]; const float* b1  = (const float*)d_in[28];
    const float* w2  = (const float*)d_in[29]; const float* b2  = (const float*)d_in[30];

    char* ws = (char*)d_ws;
    size_t off = 0;
    auto take = [&](size_t bytes) -> void* {
        void* p = ws + off;
        off = (off + bytes + 255) & ~(size_t)255;
        return p;
    };
    float*    x1u   = (float*)   take((size_t)NB * 512 * N * 4);
    float*    x2u   = (float*)   take((size_t)NB * 256 * N * 4);
    _Float16* q0    = (_Float16*)take((size_t)NB * N * 64 * 2);
    _Float16* k0    = (_Float16*)take((size_t)NB * N * 64 * 2);
    _Float16* q1    = (_Float16*)take((size_t)NB * N * 32 * 2);
    _Float16* k1    = (_Float16*)take((size_t)NB * N * 32 * 2);
    _Float16* q2    = (_Float16*)take((size_t)NB * N * 32 * 2);
    _Float16* k2    = (_Float16*)take((size_t)NB * N * 32 * 2);
    float*    u0    = (float*)   take((size_t)NB * N * 4);
    float*    u1    = (float*)   take((size_t)NB * N * 4);
    float*    u2    = (float*)   take((size_t)NB * N * 4);
    float*    aa0   = (float*)   take((size_t)NB * N * 4);
    float*    aa1   = (float*)   take((size_t)NB * N * 4);
    float*    aa2   = (float*)   take((size_t)NB * N * 4);
    float*    wvav0 = (float*)   take(512 * 4);
    float*    bva0  = (float*)   take(4);
    float*    wvav1 = (float*)   take(256 * 4);
    float*    bva1  = (float*)   take(4);
    float*    wvav2 = (float*)   take(128 * 4);
    float*    bva2  = (float*)   take(4);
    float*    w1t   = (float*)   take(896 * 32 * 4);

    // 1) upsample x2 -> 64x64 (256 ch), x1 -> 64x64 (512 ch)
    upsample_kernel<<<dim3(256 * N / 256, NB), 256, 0, stream>>>(x2, x2u, 256, 32, 32);
    upsample_kernel<<<dim3(512 * N / 256, NB), 256, 0, stream>>>(x1, x1u, 512, 16, 16);

    // 2) fold rank-1 value path + transpose w1
    prep_wvav<<<dim3(4), 128, 0, stream>>>(wa0, wv0, bv0, wvav0, bva0, 512);
    prep_wvav<<<dim3(2), 128, 0, stream>>>(wa1, wv1, bv1, wvav1, bva1, 256);
    prep_wvav<<<dim3(1), 128, 0, stream>>>(wa2, wv2, bv2, wvav2, bva2, 128);
    transpose_w1<<<dim3((32 * 896 + 255) / 256), 256, 0, stream>>>(w1, w1t);

    // 3) q/k via WMMA GEMM (f16, [N][DPAD] layout, q pre-scaled by log2e)
    proj_qk_wmma<512, 64, 64><<<dim3(N / 16, NB), 256, 0, stream>>>(
        x1u, wq0, bq0, wk0, bk0, q0, k0);
    proj_qk_wmma<256, 32, 32><<<dim3(N / 16, NB), 128, 0, stream>>>(
        x2u, wq1, bq1, wk1, bk1, q1, k1);
    proj_qk_wmma<128, 16, 32><<<dim3(N / 16, NB), 64, 0, stream>>>(
        x3, wq2, bq2, wk2, bk2, q2, k2);
    proj_u_kernel<512><<<dim3(N / 256, NB), 256, 0, stream>>>(x1u, wvav0, bva0, u0);
    proj_u_kernel<256><<<dim3(N / 256, NB), 256, 0, stream>>>(x2u, wvav1, bva1, u1);
    proj_u_kernel<128><<<dim3(N / 256, NB), 256, 0, stream>>>(x3, wvav2, bva2, u2);

    // 4) WMMA flash attention with scalar values -> per-scale attention maps
    attn_kernel<64><<<dim3(32, NB), 256, 0, stream>>>(q0, k0, u0, ba0, aa0);
    attn_kernel<32><<<dim3(32, NB), 256, 0, stream>>>(q1, k1, u1, ba1, aa1);
    attn_kernel<32><<<dim3(32, NB), 256, 0, stream>>>(q2, k2, u2, ba2, aa2);

    // 5) fused gated head
    final_kernel<<<dim3(N / 8, NB), 256, 0, stream>>>(
        x3, x2u, x1u, aa0, aa1, aa2, w1t, b1, w2, b2, (float*)d_out);
}
